// FocusVisionMambaBlock_37563783971204
// MI455X (gfx1250) — compile-verified, matrix operand order checked by reading, not run on hardware
//
#include <hip/hip_runtime.h>
#include <hip/hip_bf16.h>

#define DEVINL __device__ __forceinline__

typedef __attribute__((ext_vector_type(16))) __bf16 v16bf;
typedef __attribute__((ext_vector_type(8)))  float  v8f;
typedef __attribute__((ext_vector_type(4)))  int    i32x4;

union FragB { v16bf v; unsigned short s[16]; uint4 q[2]; };
union FragC { v8f v; float f[8]; };

#if defined(__AMDGCN__) && __has_builtin(__builtin_amdgcn_global_load_async_to_lds_b128) && __has_builtin(__builtin_amdgcn_s_wait_asynccnt)
#define USE_ASYNC_LDS 1
#else
#define USE_ASYNC_LDS 0
#endif

#if USE_ASYNC_LDS
DEVINL void async_b128(const void* g, void* l){
  __builtin_amdgcn_global_load_async_to_lds_b128(
      (__attribute__((address_space(1))) i32x4*)(g),
      (__attribute__((address_space(3))) i32x4*)(l), 0, 0);
}
#endif

DEVINL unsigned short f2bf(float f){
  unsigned int u = __float_as_uint(f);
  u += 0x7FFFu + ((u >> 16) & 1u);   // round-to-nearest-even
  return (unsigned short)(u >> 16);
}
DEVINL float bf2f(unsigned short s){ return __uint_as_float(((unsigned int)s) << 16); }
#if defined(__AMDGCN__) && __has_builtin(__builtin_amdgcn_rcpf)
DEVINL float fast_rcp(float x){ return __builtin_amdgcn_rcpf(x); }
#else
DEVINL float fast_rcp(float x){ return 1.0f / x; }
#endif
DEVINL float sigmoidf_(float x){ return fast_rcp(1.0f + __expf(-x)); }
DEVINL float siluf_(float x){ return x * sigmoidf_(x); }

DEVINL v8f wmma_bf16(const FragB& a, const FragB& b, v8f c){
  return __builtin_amdgcn_wmma_f32_16x16x32_bf16(false, a.v, false, b.v, (short)0, c, false, false);
}

// A fragment: lane holds row m=lane&15; K split per ISA 16-bit A layout:
//   lanes 0-15: K = k0+[0..7] and k0+[16..23]; lanes 16-31: K = k0+[8..15], k0+[24..31]
DEVINL void load_afrag(const unsigned short* t, int ld, int k0, int lane, FragB& f){
  const int m = lane & 15, hi = (lane >> 4) & 1;
  const unsigned short* r = t + m * ld + k0 + hi * 8;
#pragma unroll
  for (int j = 0; j < 8; ++j){ f.s[j] = r[j]; f.s[8 + j] = r[16 + j]; }
}
// B fragment: pre-packed in memory, 32 lanes x 16 halves contiguous per fragment
DEVINL void load_bfrag(const unsigned short* pk, int fragIdx, int lane, FragB& f){
  const uint4* p = (const uint4*)(pk + ((size_t)fragIdx * 32 + lane) * 16);
  f.q[0] = p[0]; f.q[1] = p[1];
}

static constexpr int Bn = 8, Cc = 96, Hh = 128, Ww = 128;
static constexpr int HW = Hh * Ww;
static constexpr int Ntok = Bn * HW;          // 131072
static constexpr int Di  = 192;
static constexpr int NTAP = 49;

// ---------------------------------------------------------------------------
// Weight pack: W (O x K) f32 row-major  ->  bf16 WMMA B fragments.
__global__ void k_pack(const float* __restrict__ Wm, unsigned short* __restrict__ dst,
                       int K, int nchunks){
  int t = blockIdx.x / nchunks, ck = blockIdx.x % nchunks;
  int lane = threadIdx.x;
  int n = lane & 15, hi = lane >> 4;
  unsigned short* o = dst + ((size_t)blockIdx.x * 32 + lane) * 16;
  const float* row = Wm + (size_t)(t * 16 + n) * K + ck * 32 + hi * 8;
#pragma unroll
  for (int j = 0; j < 8; ++j){ o[j] = f2bf(row[j]); o[8 + j] = f2bf(row[16 + j]); }
}

// ---------------------------------------------------------------------------
// ASRAM step 1: channel-wise max & mean of f*sal over HxW. One block per (b,c).
__global__ void k_reduce(const float* __restrict__ f, const float* __restrict__ sal,
                         float* __restrict__ mx, float* __restrict__ av){
  __shared__ float smax[256], ssum[256];
  int b = blockIdx.x / Cc, c = blockIdx.x % Cc, tid = threadIdx.x;
  const float* fp = f + ((size_t)(b * Cc + c)) * HW;
  const float* sp = sal + (size_t)b * HW;
  float vmax = -3.402823e38f, vsum = 0.f;
  for (int i = tid; i < HW; i += 256){
    float v = fp[i] * sp[i];
    vmax = fmaxf(vmax, v); vsum += v;
  }
  smax[tid] = vmax; ssum[tid] = vsum;
  __syncthreads();
  for (int s = 128; s > 0; s >>= 1){
    if (tid < s){ smax[tid] = fmaxf(smax[tid], smax[tid + s]); ssum[tid] += ssum[tid + s]; }
    __syncthreads();
  }
  if (tid == 0){ mx[blockIdx.x] = smax[0]; av[blockIdx.x] = ssum[0] / (float)HW; }
}

// ASRAM step 2: channel attention ca = sigmoid(mlp(mx)+mlp(av)). R = 6.
__global__ void k_ca(const float* __restrict__ mx, const float* __restrict__ av,
                     const float* __restrict__ w1, const float* __restrict__ w2,
                     float* __restrict__ ca){
  int idx = blockIdx.x * blockDim.x + threadIdx.x;
  if (idx >= Bn * Cc) return;
  int b = idx / Cc, c = idx % Cc;
  float om = 0.f, oa = 0.f;
#pragma unroll
  for (int r = 0; r < 6; ++r){
    float hm = 0.f, ha = 0.f;
    for (int k = 0; k < Cc; ++k){
      float w = w1[r * Cc + k];
      hm += w * mx[b * Cc + k];
      ha += w * av[b * Cc + k];
    }
    float w2v = w2[c * 6 + r];
    om += w2v * fmaxf(hm, 0.f);
    oa += w2v * fmaxf(ha, 0.f);
  }
  ca[idx] = sigmoidf_(om + oa);
}

// tap index -> spatial offset.  0..24: dense 5x5; then 8-tap rings for dil 4,8,16
DEVINL void tap_off(int t, int& dy, int& dx){
  if (t < 25){ dy = t / 5 - 2; dx = t % 5 - 2; }
  else { int g = (t - 25) / 8, i = (t - 25) % 8; int cell = (i < 4) ? i : i + 1;
         int d = 4 << g; dy = (cell / 3 - 1) * d; dx = (cell % 3 - 1) * d; }
}

// ASRAM step 3: fold all pointwise/diagonal weights + dw-conv kernels into
// one effective 49-tap-per-channel weight table  w_eff[tap][c].
__global__ void k_weff(const float* __restrict__ ms1, const float* __restrict__ ms3,
                       const float* __restrict__ ms5, const float* __restrict__ d4,
                       const float* __restrict__ d8, const float* __restrict__ d16,
                       const float* __restrict__ msf_dw, const float* __restrict__ msf_pw,
                       const float* __restrict__ dlf_dw, const float* __restrict__ dlf_pw,
                       const float* __restrict__ sp_dw, const float* __restrict__ sp_pw,
                       float* __restrict__ weff){
  __shared__ float spw[Cc];
  __shared__ float alpha[3 * Cc], beta[3 * Cc];
  int tid = threadIdx.x;
  if (tid < Cc) spw[tid] = sp_pw[tid] * sp_dw[tid];
  __syncthreads();
  for (int k = tid; k < 3 * Cc; k += blockDim.x){
    float am = 0.f, ad = 0.f;
    for (int c = 0; c < Cc; ++c){
      am += spw[c] * msf_pw[c * (3 * Cc) + k];
      ad += spw[c] * dlf_pw[c * (3 * Cc) + k];
    }
    alpha[k] = am * msf_dw[k];
    beta[k]  = ad * dlf_dw[k];
  }
  __syncthreads();
  for (int i = tid; i < NTAP * Cc; i += blockDim.x){
    int t = i / Cc, c = i % Cc;
    float w = 0.f;
    if (t < 25){
      int dy = t / 5 - 2, dx = t % 5 - 2;
      w = alpha[2 * Cc + c] * ms5[c * 25 + t];
      if (dy >= -1 && dy <= 1 && dx >= -1 && dx <= 1)
        w += alpha[Cc + c] * ms3[c * 9 + (dy + 1) * 3 + (dx + 1)];
      if (t == 12){
        w += alpha[c] * ms1[c];
        w += beta[c] * d4[c * 9 + 4] + beta[Cc + c] * d8[c * 9 + 4]
           + beta[2 * Cc + c] * d16[c * 9 + 4];
      }
    } else {
      int g = (t - 25) / 8, i8 = (t - 25) % 8; int cell = (i8 < 4) ? i8 : i8 + 1;
      const float* dw = (g == 0) ? d4 : (g == 1) ? d8 : d16;
      w = beta[g * Cc + c] * dw[c * 9 + cell];
    }
    weff[t * Cc + c] = w;
  }
}

// ASRAM step 4: spatial attention map. One thread per pixel, 49 taps x 96 channels.
__global__ void __launch_bounds__(256) k_sa(const float* __restrict__ f,
                                            const float* __restrict__ weff,
                                            float* __restrict__ sa){
  __shared__ float wl[NTAP * Cc];
  int tid = threadIdx.x;
  for (int i = tid; i < NTAP * Cc; i += 256) wl[i] = weff[i];
  __syncthreads();
  int idx = blockIdx.x * 256 + tid;
  int b = idx / HW, p = idx % HW;
  int h = p / Ww, w = p % Ww;
  float acc = 0.f;
  const float* fb = f + (size_t)b * Cc * HW;
  for (int t = 0; t < NTAP; ++t){
    int dy, dx; tap_off(t, dy, dx);
    int hh = h + dy, ww = w + dx;
    if (hh < 0 || hh >= Hh || ww < 0 || ww >= Ww) continue;
    const float* fp = fb + hh * Ww + ww;
    const float* wp = wl + t * Cc;
#pragma unroll 4
    for (int c = 0; c < Cc; ++c) acc += wp[c] * fp[(size_t)c * HW];
  }
  sa[idx] = sigmoidf_(acc);
}

// ASRAM step 5: y = f*(1 + ca*sa), written NHWC f32 (VSS token layout).
__global__ void k_combine(const float* __restrict__ f, const float* __restrict__ ca,
                          const float* __restrict__ sa, float* __restrict__ xnhwc){
  size_t idx = (size_t)blockIdx.x * 256 + threadIdx.x;   // NCHW order (coalesced read)
  int w = idx % Ww; int h = (idx / Ww) % Hh; int c = (idx / HW) % Cc; int b = idx / ((size_t)Cc * HW);
  float v = f[idx] * (1.f + ca[b * Cc + c] * sa[(size_t)b * HW + h * Ww + w]);
  xnhwc[(((size_t)b * Hh + h) * Ww + w) * Cc + c] = v;
}

// ---------------------------------------------------------------------------
// VSS kernel 1: LN over C, then in_proj GEMM (96 -> 384) via WMMA bf16.
// Block = 256 thr (8 waves), 16 tokens; each wave: 3 interleaved col tiles.
__global__ void __launch_bounds__(256) k_ln_inproj(
    const float* __restrict__ x, const float* __restrict__ g, const float* __restrict__ bb,
    const unsigned short* __restrict__ wpack,
    unsigned short* __restrict__ xpart, unsigned short* __restrict__ siluz){
  __shared__ float xin[16 * Cc];
  __shared__ unsigned short lnt[16 * Cc];
  __shared__ float mean[16], rstd[16];
  int tid = threadIdx.x;
  int tok0 = blockIdx.x * 16;
  size_t base = (size_t)tok0 * Cc;
#if USE_ASYNC_LDS
  for (int ch = tid; ch < (16 * Cc) / 4; ch += 256)
    async_b128(x + base + ch * 4, &xin[ch * 4]);
  __builtin_amdgcn_s_wait_asynccnt(0);
#else
  for (int i = tid; i < 16 * Cc; i += 256) xin[i] = x[base + i];
#endif
  __syncthreads();
  if (tid < 16){
    float s = 0.f, s2 = 0.f;
    for (int k = 0; k < Cc; ++k){ float v = xin[tid * Cc + k]; s += v; s2 += v * v; }
    float m = s / (float)Cc, var = s2 / (float)Cc - m * m;
    mean[tid] = m; rstd[tid] = rsqrtf(var + 1e-5f);
  }
  __syncthreads();
  for (int i = tid; i < 16 * Cc; i += 256){
    int t = i / Cc, ch = i % Cc;
    lnt[i] = f2bf((xin[i] - mean[t]) * rstd[t] * g[ch] + bb[ch]);
  }
  __syncthreads();
  int wv = tid >> 5, lane = tid & 31;
  FragB a[3];
#pragma unroll
  for (int c = 0; c < 3; ++c) load_afrag(lnt, Cc, c * 32, lane, a[c]);
  FragC acc[3];
#pragma unroll
  for (int t = 0; t < 3; ++t)
#pragma unroll
    for (int i = 0; i < 8; ++i) acc[t].f[i] = 0.f;
  // 3 independent accumulator chains -> hides WMMA->WMMA RAW hazard
#pragma unroll
  for (int c = 0; c < 3; ++c){
    FragB bf[3];
#pragma unroll
    for (int t = 0; t < 3; ++t) load_bfrag(wpack, (wv * 3 + t) * 3 + c, lane, bf[t]);
#pragma unroll
    for (int t = 0; t < 3; ++t) acc[t].v = wmma_bf16(a[c], bf[t], acc[t].v);
  }
  int hi = lane >> 4;
#pragma unroll
  for (int tt = 0; tt < 3; ++tt){
    int tile = wv * 3 + tt;
    int n = tile * 16 + (lane & 15);
    if (tile < 12){               // wave-uniform split: x_part vs silu(z)
#pragma unroll
      for (int r = 0; r < 8; ++r){
        size_t tok = (size_t)tok0 + r + 8 * hi;
        xpart[tok * Di + n] = f2bf(acc[tt].f[r]);
      }
    } else {
#pragma unroll
      for (int r = 0; r < 8; ++r){
        size_t tok = (size_t)tok0 + r + 8 * hi;
        siluz[tok * Di + (n - Di)] = f2bf(siluf_(acc[tt].f[r]));
      }
    }
  }
}

// VSS kernel 2: depthwise 3x3 conv (NHWC) + bias, silu, + dt_bias.
__global__ void k_dwconv(const unsigned short* __restrict__ xp,
                         const float* __restrict__ cw, const float* __restrict__ cb,
                         const float* __restrict__ dtb, unsigned short* __restrict__ xs){
  size_t idx = (size_t)blockIdx.x * 256 + threadIdx.x;
  int c = idx % Di; size_t pix = idx / Di;
  int w = pix % Ww; int h = (pix / Ww) % Hh; int b = pix / ((size_t)Hh * Ww);
  float acc = 0.f;
#pragma unroll
  for (int ky = 0; ky < 3; ++ky){
    int hh = h + ky - 1; if (hh < 0 || hh >= Hh) continue;
#pragma unroll
    for (int kx = 0; kx < 3; ++kx){
      int ww = w + kx - 1; if (ww < 0 || ww >= Ww) continue;
      acc += bf2f(xp[(((size_t)b * Hh + hh) * Ww + ww) * Di + c]) * cw[c * 9 + ky * 3 + kx];
    }
  }
  acc += cb[c];
  xs[idx] = f2bf(siluf_(acc) + dtb[c]);
}

// VSS kernel 3: fused  h=silu(xs@w1^T) -> y=h@w2^T -> LN(Di) -> *silu(z)
//               -> out=@out_proj^T -> x += out.
// Block = 96 thr (3 waves), 16 tokens; 4 interleaved tiles/wave in GEMM1/2.
__global__ void __launch_bounds__(96) k_mamba(
    const unsigned short* __restrict__ xs, const unsigned short* __restrict__ sz,
    const unsigned short* __restrict__ w1p, const unsigned short* __restrict__ w2p,
    const unsigned short* __restrict__ wop,
    const float* __restrict__ ng, const float* __restrict__ nb,
    float* __restrict__ xres){
  __shared__ unsigned short tA[16 * Di];
  __shared__ unsigned short tB[16 * Di];
  __shared__ float tY[16 * Di];
  __shared__ float mean[16], rstd[16];
  int tid = threadIdx.x;
  size_t base = (size_t)blockIdx.x * 16 * Di;
#if USE_ASYNC_LDS
  for (int ch = tid; ch < (16 * Di) / 8; ch += 96)
    async_b128(xs + base + ch * 8, &tA[ch * 8]);
  __builtin_amdgcn_s_wait_asynccnt(0);
#else
  for (int i = tid; i < 16 * Di; i += 96) tA[i] = xs[base + i];
#endif
  __syncthreads();
  int wv = tid / 32, lane = tid % 32;
  int hi = lane >> 4;
  // GEMM1: h = silu(xs @ w1^T); wave wv -> tiles wv*4 .. wv*4+3 (4 indep chains)
  {
    FragB a[6];
#pragma unroll
    for (int c = 0; c < 6; ++c) load_afrag(tA, Di, c * 32, lane, a[c]);
    FragC acc[4];
#pragma unroll
    for (int t = 0; t < 4; ++t)
#pragma unroll
      for (int i = 0; i < 8; ++i) acc[t].f[i] = 0.f;
#pragma unroll
    for (int c = 0; c < 6; ++c){
      FragB bf[4];
#pragma unroll
      for (int t = 0; t < 4; ++t) load_bfrag(w1p, (wv * 4 + t) * 6 + c, lane, bf[t]);
#pragma unroll
      for (int t = 0; t < 4; ++t) acc[t].v = wmma_bf16(a[c], bf[t], acc[t].v);
    }
#pragma unroll
    for (int tt = 0; tt < 4; ++tt){
      int n = (wv * 4 + tt) * 16 + (lane & 15);
#pragma unroll
      for (int r = 0; r < 8; ++r) tB[(r + 8 * hi) * Di + n] = f2bf(siluf_(acc[tt].f[r]));
    }
  }
  __syncthreads();
  // GEMM2: y = h @ w2^T (f32 into LDS for LN)
  {
    FragB a[6];
#pragma unroll
    for (int c = 0; c < 6; ++c) load_afrag(tB, Di, c * 32, lane, a[c]);
    FragC acc[4];
#pragma unroll
    for (int t = 0; t < 4; ++t)
#pragma unroll
      for (int i = 0; i < 8; ++i) acc[t].f[i] = 0.f;
#pragma unroll
    for (int c = 0; c < 6; ++c){
      FragB bf[4];
#pragma unroll
      for (int t = 0; t < 4; ++t) load_bfrag(w2p, (wv * 4 + t) * 6 + c, lane, bf[t]);
#pragma unroll
      for (int t = 0; t < 4; ++t) acc[t].v = wmma_bf16(a[c], bf[t], acc[t].v);
    }
#pragma unroll
    for (int tt = 0; tt < 4; ++tt){
      int n = (wv * 4 + tt) * 16 + (lane & 15);
#pragma unroll
      for (int r = 0; r < 8; ++r) tY[(r + 8 * hi) * Di + n] = acc[tt].f[r];
    }
  }
  __syncthreads();
  if (tid < 16){
    float s = 0.f, s2 = 0.f;
    for (int k = 0; k < Di; ++k){ float v = tY[tid * Di + k]; s += v; s2 += v * v; }
    float m = s / (float)Di, var = s2 / (float)Di - m * m;
    mean[tid] = m; rstd[tid] = rsqrtf(var + 1e-5f);
  }
  __syncthreads();
  for (int i = tid; i < 16 * Di; i += 96){
    int t = i / Di, ch = i % Di;
    float v = (tY[i] - mean[t]) * rstd[t] * ng[ch] + nb[ch];
    v *= bf2f(sz[base + i]);        // silu(z) pre-applied at store time
    tA[i] = f2bf(v);
  }
  __syncthreads();
  // GEMM3: out_proj (192 -> 96), residual add; wave wv -> tiles wv*2, wv*2+1
  {
    FragB a[6];
#pragma unroll
    for (int c = 0; c < 6; ++c) load_afrag(tA, Di, c * 32, lane, a[c]);
    FragC acc[2];
#pragma unroll
    for (int t = 0; t < 2; ++t)
#pragma unroll
      for (int i = 0; i < 8; ++i) acc[t].f[i] = 0.f;
#pragma unroll
    for (int c = 0; c < 6; ++c){
      FragB bf[2];
#pragma unroll
      for (int t = 0; t < 2; ++t) load_bfrag(wop, (wv * 2 + t) * 6 + c, lane, bf[t]);
#pragma unroll
      for (int t = 0; t < 2; ++t) acc[t].v = wmma_bf16(a[c], bf[t], acc[t].v);
    }
    size_t tb = (size_t)blockIdx.x * 16 * Cc;
#pragma unroll
    for (int tt = 0; tt < 2; ++tt){
      int n = (wv * 2 + tt) * 16 + (lane & 15);
#pragma unroll
      for (int r = 0; r < 8; ++r){
        size_t o = tb + (size_t)(r + 8 * hi) * Cc + n;
        xres[o] += acc[tt].f[r];
      }
    }
  }
}

// NHWC -> NCHW final transpose.
__global__ void k_out(const float* __restrict__ xnhwc, float* __restrict__ out){
  size_t idx = (size_t)blockIdx.x * 256 + threadIdx.x;     // NCHW order (coalesced write)
  int w = idx % Ww; int h = (idx / Ww) % Hh; int c = (idx / HW) % Cc; int b = idx / ((size_t)Cc * HW);
  out[idx] = xnhwc[(((size_t)b * Hh + h) * Ww + w) * Cc + c];
}

// ---------------------------------------------------------------------------
extern "C" void kernel_launch(void* const* d_in, const int* in_sizes, int n_in,
                              void* d_out, int out_size, void* d_ws, size_t ws_size,
                              hipStream_t stream) {
  (void)in_sizes; (void)n_in; (void)out_size; (void)ws_size;
  const float* f   = (const float*)d_in[0];
  const float* sal = (const float*)d_in[1];
  const float* scam_w1 = (const float*)d_in[2];
  const float* scam_w2 = (const float*)d_in[3];
  const float* ms_w1 = (const float*)d_in[4];
  const float* ms_w3 = (const float*)d_in[5];
  const float* ms_w5 = (const float*)d_in[6];
  const float* dil4  = (const float*)d_in[7];
  const float* dil8  = (const float*)d_in[8];
  const float* dil16 = (const float*)d_in[9];
  const float* msf_dw = (const float*)d_in[10];
  const float* msf_pw = (const float*)d_in[11];
  const float* dlf_dw = (const float*)d_in[12];
  const float* dlf_pw = (const float*)d_in[13];
  const float* sp_dw  = (const float*)d_in[14];
  const float* sp_pw  = (const float*)d_in[15];

  size_t off = 0;
  auto alloc = [&](size_t bytes) -> void* {
    void* p = (char*)d_ws + off;
    off += (bytes + 255) & ~(size_t)255;
    return p;
  };
  float* buf_x = (float*)alloc((size_t)Ntok * Cc * 4);           // NHWC activations
  unsigned short* buf_a = (unsigned short*)alloc((size_t)Ntok * Di * 2);
  unsigned short* buf_b = (unsigned short*)alloc((size_t)Ntok * Di * 2);
  unsigned short* buf_c = (unsigned short*)alloc((size_t)Ntok * Di * 2);
  float* sa_m  = (float*)alloc((size_t)Ntok * 4);
  float* ca_m  = (float*)alloc(Bn * Cc * 4);
  float* mx_m  = (float*)alloc(Bn * Cc * 4);
  float* av_m  = (float*)alloc(Bn * Cc * 4);
  float* weff  = (float*)alloc(NTAP * Cc * 4);
  unsigned short* p_in[2], *p_w1[2], *p_w2[2], *p_op[2];
  for (int v = 0; v < 2; ++v){
    p_in[v] = (unsigned short*)alloc((size_t)384 * 96 * 2);
    p_w1[v] = (unsigned short*)alloc((size_t)192 * 192 * 2);
    p_w2[v] = (unsigned short*)alloc((size_t)192 * 192 * 2);
    p_op[v] = (unsigned short*)alloc((size_t)96 * 192 * 2);
  }

  // Pack all GEMM weights -> bf16 WMMA fragments
  for (int v = 0; v < 2; ++v){
    int pb = 16 + v * 11;
    k_pack<<<24 * 3, 32, 0, stream>>>((const float*)d_in[pb + 2], p_in[v], 96, 3);   // in_proj (384x96)
    k_pack<<<12 * 6, 32, 0, stream>>>((const float*)d_in[pb + 5], p_w1[v], 192, 6);  // mamba_w1
    k_pack<<<12 * 6, 32, 0, stream>>>((const float*)d_in[pb + 6], p_w2[v], 192, 6);  // mamba_w2
    k_pack<<< 6 * 6, 32, 0, stream>>>((const float*)d_in[pb + 9], p_op[v], 192, 6);  // out_proj (96x192)
  }

  // ASRAM
  k_reduce<<<Bn * Cc, 256, 0, stream>>>(f, sal, mx_m, av_m);
  k_ca<<<3, 256, 0, stream>>>(mx_m, av_m, scam_w1, scam_w2, ca_m);
  k_weff<<<1, 256, 0, stream>>>(ms_w1, ms_w3, ms_w5, dil4, dil8, dil16,
                                msf_dw, msf_pw, dlf_dw, dlf_pw, sp_dw, sp_pw, weff);
  k_sa<<<Ntok / 256, 256, 0, stream>>>(f, weff, sa_m);
  k_combine<<<(Ntok * Cc) / 256, 256, 0, stream>>>(f, ca_m, sa_m, buf_x);

  // Two VSS blocks
  for (int v = 0; v < 2; ++v){
    int pb = 16 + v * 11;
    const float* ln_g   = (const float*)d_in[pb + 0];
    const float* ln_b   = (const float*)d_in[pb + 1];
    const float* conv_w = (const float*)d_in[pb + 3];
    const float* conv_b = (const float*)d_in[pb + 4];
    const float* norm_g = (const float*)d_in[pb + 7];
    const float* norm_b = (const float*)d_in[pb + 8];
    const float* dt_b   = (const float*)d_in[pb + 10];

    k_ln_inproj<<<Ntok / 16, 256, 0, stream>>>(buf_x, ln_g, ln_b, p_in[v], buf_a, buf_b);
    k_dwconv<<<(Ntok * Di) / 256, 256, 0, stream>>>(buf_a, conv_w, conv_b, dt_b, buf_c);
    k_mamba<<<Ntok / 16, 96, 0, stream>>>(buf_c, buf_b, p_w1[v], p_w2[v], p_op[v],
                                          norm_g, norm_b, buf_x);
  }

  k_out<<<(Ntok * Cc) / 256, 256, 0, stream>>>(buf_x, (float*)d_out);
}
